// GraphPeakSelector_4844723110435
// MI455X (gfx1250) — matrix operand order repaired and checked by reading, not compiled, not run on hardware
//
#include <hip/hip_runtime.h>
#include <hip/hip_bf16.h>
#include <math.h>

typedef __attribute__((ext_vector_type(16))) _Float16 v16h;
typedef __attribute__((ext_vector_type(8)))  float    v8f;

#define H_MAP 1024
#define W_MAP 1024
#define NMAPS 16
#define TOPK 8
#define VIEWS 16
#define CFEAT 128
#define DEMB 64
#define HF 32
#define WF 32

// Output tuple layout (flat, in return order)
#define OFF_RW  0            // reweighted: 16*1024*1024
#define OFF_CNE 16777216     // curr_node_embed: 64
#define OFF_PC  16777280     // peak_coords: 16*8*2
#define OFF_TS  16777536     // transition_scores: 16*8
#define OFF_PNE 16777664     // peak_node_embeds: 16*8*64

// ---------------------------------------------------------------------------
// Kernel 1: 5x5 maxpool NMS per 32x32 tile (with 2-halo in LDS), per-tile top-8
// ---------------------------------------------------------------------------
__global__ void k_nms_tile(const float* __restrict__ heat,
                           float* __restrict__ candVal, int* __restrict__ candIdx) {
  __shared__ float tile[36 * 36];
  __shared__ float cval[1024];
  __shared__ int   cidx[1024];
  __shared__ int   scnt;
  __shared__ float rval[256];
  __shared__ int   rslot[256];

  const int tid = threadIdx.x;
  const int map = blockIdx.z;
  const int bx = blockIdx.x * 32, by = blockIdx.y * 32;
  const float* hm = heat + (size_t)map * H_MAP * W_MAP;

  for (int i = tid; i < 36 * 36; i += 256) {
    int ly = i / 36, lx = i - ly * 36;
    int gy = by + ly - 2, gx = bx + lx - 2;
    float v = -INFINITY;
    if (gy >= 0 && gy < H_MAP && gx >= 0 && gx < W_MAP) v = hm[(size_t)gy * W_MAP + gx];
    tile[i] = v;
  }
  if (tid == 0) scnt = 0;
  __syncthreads();

  for (int i = tid; i < 1024; i += 256) {
    int py = i >> 5, px = i & 31;
    float c = tile[(py + 2) * 36 + (px + 2)];
    float wmax = -INFINITY;
#pragma unroll
    for (int dy = 0; dy < 5; ++dy)
#pragma unroll
      for (int dx = 0; dx < 5; ++dx)
        wmax = fmaxf(wmax, tile[(py + dy) * 36 + (px + dx)]);
    if (c >= wmax && c > 0.0f) {   // surviving positive NMS peak
      int pos = atomicAdd(&scnt, 1);
      if (pos < 1024) { cval[pos] = c; cidx[pos] = (by + py) * W_MAP + (bx + px); }
    }
  }
  __syncthreads();

  int n = scnt < 1024 ? scnt : 1024;
  const int tileId = blockIdx.y * 32 + blockIdx.x;
  float* outV = candVal + ((size_t)map * 1024 + tileId) * TOPK;
  int*   outI = candIdx + ((size_t)map * 1024 + tileId) * TOPK;

  for (int j = 0; j < TOPK; ++j) {
    float bv = -INFINITY; int bs = -1;
    for (int i = tid; i < n; i += 256)
      if (cval[i] > bv) { bv = cval[i]; bs = i; }
    rval[tid] = bv; rslot[tid] = bs;
    __syncthreads();
    for (int s = 128; s > 0; s >>= 1) {
      if (tid < s && rval[tid + s] > rval[tid]) { rval[tid] = rval[tid + s]; rslot[tid] = rslot[tid + s]; }
      __syncthreads();
    }
    if (tid == 0) {
      int slot = rslot[0];
      if (slot >= 0) { outV[j] = rval[0]; outI[j] = cidx[slot]; cval[slot] = -INFINITY; }
      else           { outV[j] = -INFINITY; outI[j] = 0; }
    }
    __syncthreads();
  }
}

// ---------------------------------------------------------------------------
// Kernel 2: per-map reduction of 1024*8 candidates to global top-8 (sorted desc)
// ---------------------------------------------------------------------------
__global__ void k_topk_final(const float* __restrict__ candVal, const int* __restrict__ candIdx,
                             int* __restrict__ wsRows, int* __restrict__ wsCols,
                             float* __restrict__ out) {
  __shared__ float vals[8192];
  __shared__ float rv[256];
  __shared__ int   rs[256];
  const int tid = threadIdx.x;
  const int map = blockIdx.x;
  for (int i = tid; i < 8192; i += 256) vals[i] = candVal[(size_t)map * 8192 + i];
  __syncthreads();

  for (int j = 0; j < TOPK; ++j) {
    float bv = -INFINITY; int bs = -1;
    for (int i = tid; i < 8192; i += 256)
      if (vals[i] > bv) { bv = vals[i]; bs = i; }
    rv[tid] = bv; rs[tid] = bs;
    __syncthreads();
    for (int s = 128; s > 0; s >>= 1) {
      if (tid < s && rv[tid + s] > rv[tid]) { rv[tid] = rv[tid + s]; rs[tid] = rs[tid + s]; }
      __syncthreads();
    }
    if (tid == 0) {
      int slot = rs[0];
      int gi = 0;
      if (slot >= 0 && rv[0] > -INFINITY) { gi = candIdx[(size_t)map * 8192 + slot]; vals[slot] = -INFINITY; }
      int row = gi >> 10, col = gi & 1023;
      wsRows[map * TOPK + j] = row;
      wsCols[map * TOPK + j] = col;
      out[OFF_PC + (map * TOPK + j) * 2 + 0] = (float)row;
      out[OFF_PC + (map * TOPK + j) * 2 + 1] = (float)col;
    }
    __syncthreads();
  }
}

// ---------------------------------------------------------------------------
// Kernel 3: pooled channel mean of feature_map (16,128,32,32) -> (128,)
// ---------------------------------------------------------------------------
__global__ void k_pooled_mean(const float* __restrict__ fm, float* __restrict__ pooled) {
  __shared__ float red[256];
  const int c = blockIdx.x;
  const int tid = threadIdx.x;
  float s = 0.f;
  for (int v = 0; v < VIEWS; ++v) {
    const float* base = fm + ((size_t)v * CFEAT + c) * (HF * WF);
    for (int i = tid; i < HF * WF; i += 256) s += base[i];
  }
  red[tid] = s;
  __syncthreads();
  for (int st = 128; st > 0; st >>= 1) {
    if (tid < st) red[tid] += red[tid + st];
    __syncthreads();
  }
  if (tid == 0) pooled[c] = red[0] * (1.0f / (VIEWS * HF * WF));
}

// ---------------------------------------------------------------------------
// WMMA fragment builders (wave32, f16 16x16x32; layouts per CDNA5 ISA 7.12.2)
// ---------------------------------------------------------------------------
__device__ inline v16h frag_a_lds(const _Float16* A, int lda, int m0, int k0, int lane) {
  v16h a;
  const int m = m0 + (lane & 15);
  const int khi = (lane >> 4) << 3;   // lanes 16-31: +8
#pragma unroll
  for (int h = 0; h < 16; ++h) {
    int k = k0 + (h & 7) + ((h >> 3) << 4) + khi;
    a[h] = A[(size_t)m * lda + k];
  }
  return a;
}
__device__ inline v16h frag_b_glb(const float* B, int ldb, int k0, int n0, int lane) {
  v16h b;
  const int n = n0 + (lane & 15);
  const int khi = (lane >> 4) << 3;   // lanes 16-31: +8
#pragma unroll
  for (int h = 0; h < 16; ++h) {
    int k = k0 + (h & 7) + ((h >> 3) << 4) + khi;
    b[h] = (_Float16)B[(size_t)k * ldb + n];
  }
  return b;
}

// ---------------------------------------------------------------------------
// Kernel 4 (single workgroup, 128 threads = 4 waves):
//   bilinear sample -> MLP1(WMMA) -> MLP2(WMMA) -> L2 norm -> curr embed ->
//   context/proto -> transition scores -> softmax weights
// ---------------------------------------------------------------------------
__global__ void k_fused_embed(const float* __restrict__ fm, const float* __restrict__ prev,
                              const float* __restrict__ n1w, const float* __restrict__ n1b,
                              const float* __restrict__ n2w, const float* __restrict__ n2b,
                              const float* __restrict__ p1w, const float* __restrict__ p1b,
                              const float* __restrict__ p2w, const float* __restrict__ p2b,
                              const float* __restrict__ cw,  const float* __restrict__ cb,
                              const float* __restrict__ sw,  const float* __restrict__ sb,
                              const int* __restrict__ wsRows, const int* __restrict__ wsCols,
                              const float* __restrict__ pooled, float* __restrict__ wsW,
                              float* __restrict__ out) {
  __shared__ __align__(16) char smem[32768 + 16384 + 2048];
  _Float16* As = (_Float16*)smem;             // 128x128 f16 sampled (32 KB)
  float*    E  = (float*)smem;                // aliases As after GEMM1 done (128x64 f32)
  _Float16* H1 = (_Float16*)(smem + 32768);   // 128x64 f16 hidden (16 KB)
  float*    sm = (float*)(smem + 49152);      // small scratch (<=512 floats)

  const int tid  = threadIdx.x;               // 0..127
  const int lane = tid & 31;
  const int wave = tid >> 5;

  // ---- Stage 1: bilinear sample at peak coords -> As (f16) ----
  {
    const int v = tid >> 3;
    const float ys = (float)wsRows[tid] * ((float)HF / (float)H_MAP);
    const float xs = (float)wsCols[tid] * ((float)WF / (float)W_MAP);
    const float y0f = floorf(ys), x0f = floorf(xs);
    const float wy = ys - y0f, wx = xs - x0f;
    const int y0 = (int)y0f, x0 = (int)x0f;
    const float w00 = (1.f - wy) * (1.f - wx), w01 = (1.f - wy) * wx;
    const float w10 = wy * (1.f - wx),         w11 = wy * wx;
    const float* fmv = fm + (size_t)v * CFEAT * (HF * WF);
    for (int c = 0; c < CFEAT; ++c) {
      const float* fc = fmv + (size_t)c * (HF * WF);
      float acc = 0.f;
      if (y0 >= 0   && y0 < HF   && x0 >= 0   && x0 < WF)   acc += fc[y0 * WF + x0] * w00;
      if (y0 >= 0   && y0 < HF   && x0+1 >= 0 && x0+1 < WF) acc += fc[y0 * WF + x0 + 1] * w01;
      if (y0+1 >= 0 && y0+1 < HF && x0 >= 0   && x0 < WF)   acc += fc[(y0+1) * WF + x0] * w10;
      if (y0+1 >= 0 && y0+1 < HF && x0+1 >= 0 && x0+1 < WF) acc += fc[(y0+1) * WF + x0 + 1] * w11;
      As[(size_t)tid * CFEAT + c] = (_Float16)acc;
    }
  }
  __syncthreads();

  // ---- Stage 2: GEMM1 (128x128 @ 128x64) + bias + ReLU -> H1 f16 ----
  for (int t = wave; t < 32; t += 4) {      // 8 M-tiles x 4 N-tiles
    const int mt = t >> 2, nt = t & 3;
    v8f acc = {};
#pragma unroll
    for (int ks = 0; ks < 4; ++ks) {
      v16h a = frag_a_lds(As, CFEAT, mt * 16, ks * 32, lane);
      v16h b = frag_b_glb(n1w, DEMB, ks * 32, nt * 16, lane);
      acc = __builtin_amdgcn_wmma_f32_16x16x32_f16(false, a, false, b, (short)0, acc, false, false);
    }
    const int n = nt * 16 + (lane & 15);
    const float bias = n1b[n];
#pragma unroll
    for (int r = 0; r < 8; ++r) {
      const int m = mt * 16 + r + ((lane >> 4) << 3);
      H1[(size_t)m * DEMB + n] = (_Float16)fmaxf(acc[r] + bias, 0.f);
    }
  }
  __syncthreads();

  // ---- Stage 3: GEMM2 (128x64 @ 64x64) + bias -> E f32 (aliases As) ----
  for (int t = wave; t < 32; t += 4) {
    const int mt = t >> 2, nt = t & 3;
    v8f acc = {};
#pragma unroll
    for (int ks = 0; ks < 2; ++ks) {
      v16h a = frag_a_lds(H1, DEMB, mt * 16, ks * 32, lane);
      v16h b = frag_b_glb(n2w, DEMB, ks * 32, nt * 16, lane);
      acc = __builtin_amdgcn_wmma_f32_16x16x32_f16(false, a, false, b, (short)0, acc, false, false);
    }
    const int n = nt * 16 + (lane & 15);
    const float bias = n2b[n];
#pragma unroll
    for (int r = 0; r < 8; ++r) {
      const int m = mt * 16 + r + ((lane >> 4) << 3);
      E[(size_t)m * DEMB + n] = acc[r] + bias;
    }
  }
  __syncthreads();

  // ---- Stage 4: row L2-normalize -> peak_node_embeds ----
  {
    float ss = 0.f;
    for (int d = 0; d < DEMB; ++d) { float x = E[tid * DEMB + d]; ss += x * x; }
    const float inv = 1.0f / fmaxf(sqrtf(ss), 1e-12f);
    for (int d = 0; d < DEMB; ++d) {
      float x = E[tid * DEMB + d] * inv;
      E[tid * DEMB + d] = x;
      out[OFF_PNE + tid * DEMB + d] = x;
    }
  }
  __syncthreads();

  // ---- Stage 5: curr_node_embed = l2norm(relu(pooled@p1+b1)@p2+b2) ----
  if (tid < DEMB) {
    float h = p1b[tid];
    for (int j = 0; j < CFEAT; ++j) h += pooled[j] * p1w[j * DEMB + tid];
    sm[tid] = fmaxf(h, 0.f);                          // hidden
  }
  __syncthreads();
  if (tid < DEMB) {
    float c = p2b[tid];
    for (int j = 0; j < DEMB; ++j) c += sm[j] * p2w[j * DEMB + tid];
    sm[64 + tid] = c;                                 // raw curr
  }
  __syncthreads();
  if (tid < DEMB) {
    float ss = 0.f;
    for (int j = 0; j < DEMB; ++j) ss += sm[64 + j] * sm[64 + j];
    const float inv = 1.0f / fmaxf(sqrtf(ss), 1e-12f);
    const float c = sm[64 + tid] * inv;
    sm[128 + tid] = c;                                // curr_node_embed
    out[OFF_CNE + tid] = c;
  }
  __syncthreads();

  // ---- Stage 6: context/proto (identical for all views) ----
  if (tid < DEMB) {
    float h = cb[tid];
    for (int j = 0; j < 2 * DEMB; ++j) {
      const float x = (j < DEMB) ? prev[j] : sm[128 + (j - DEMB)];
      h += x * cw[j * DEMB + tid];
    }
    sm[192 + tid] = fmaxf(h, 0.f);                    // context
  }
  __syncthreads();
  if (tid < DEMB) {
    float pr = sb[tid];
    for (int j = 0; j < DEMB; ++j) pr += sm[192 + j] * sw[j * DEMB + tid];
    sm[256 + tid] = pr;                               // proto
  }
  __syncthreads();

  // ---- Stage 7: transition scores = <embed, proto> ----
  {
    float s = 0.f;
    for (int d = 0; d < DEMB; ++d) s += E[tid * DEMB + d] * sm[256 + d];
    sm[320 + tid] = s;
    out[OFF_TS + tid] = s;
  }
  __syncthreads();

  // ---- Stage 8: softmax over K -> peak_weights ----
  if (tid < VIEWS) {
    float mx = -INFINITY;
    for (int k = 0; k < TOPK; ++k) mx = fmaxf(mx, sm[320 + tid * TOPK + k]);
    float den = 0.f;
    float e[TOPK];
#pragma unroll
    for (int k = 0; k < TOPK; ++k) { e[k] = expf(sm[320 + tid * TOPK + k] - mx); den += e[k]; }
    const float inv = 1.0f / den;
#pragma unroll
    for (int k = 0; k < TOPK; ++k) wsW[tid * TOPK + k] = e[k] * inv;
  }
}

// ---------------------------------------------------------------------------
// Kernel 5: separable gaussian bias + heatmap add (the HBM-bound stage)
// ---------------------------------------------------------------------------
__global__ void k_bias_add(const float* __restrict__ heat,
                           const int* __restrict__ wsRows, const int* __restrict__ wsCols,
                           const float* __restrict__ wsW, float* __restrict__ out) {
  __shared__ float gcs[TOPK * 1024];
  __shared__ float aw[TOPK];
  __shared__ int   rk[TOPK];
  __shared__ int   ck[TOPK];

  const int v = blockIdx.y;
  const int chunk = blockIdx.x;         // 64 chunks of 16 rows
  const int tid = threadIdx.x;
  const float inv2s2 = 1.0f / 18.0f;    // 1/(2*sigma^2), sigma=3

  if (tid < TOPK) {
    aw[tid] = wsW[v * TOPK + tid];
    rk[tid] = wsRows[v * TOPK + tid];
    ck[tid] = wsCols[v * TOPK + tid];
  }
  __syncthreads();
  for (int i = tid; i < TOPK * 1024; i += 256) {
    const int k = i >> 10, c = i & 1023;
    const float d = (float)c - (float)ck[k];
    gcs[i] = expf(-d * d * inv2s2);
  }
  __syncthreads();

  const size_t base = (size_t)v * H_MAP * W_MAP;
  for (int rr = 0; rr < 16; ++rr) {
    const int r = chunk * 16 + rr;
    float grk[TOPK];
#pragma unroll
    for (int k = 0; k < TOPK; ++k) {
      const float d = (float)r - (float)rk[k];
      grk[k] = aw[k] * expf(-d * d * inv2s2);
    }
    const float4* src = (const float4*)(heat + base + (size_t)r * W_MAP);
    float4*       dst = (float4*)(out + OFF_RW + base + (size_t)r * W_MAP);
    if (rr + 1 < 16)
      __builtin_prefetch(heat + base + (size_t)(r + 1) * W_MAP + tid * 4, 0, 1);
    float4 h = src[tid];
    const int c0 = tid * 4;
    float b0 = 0.f, b1 = 0.f, b2 = 0.f, b3 = 0.f;
#pragma unroll
    for (int k = 0; k < TOPK; ++k) {
      const float* g = &gcs[k << 10];
      b0 += grk[k] * g[c0 + 0];
      b1 += grk[k] * g[c0 + 1];
      b2 += grk[k] * g[c0 + 2];
      b3 += grk[k] * g[c0 + 3];
    }
    h.x += b0; h.y += b1; h.z += b2; h.w += b3;
    dst[tid] = h;
  }
}

// ---------------------------------------------------------------------------
extern "C" void kernel_launch(void* const* d_in, const int* in_sizes, int n_in,
                              void* d_out, int out_size, void* d_ws, size_t ws_size,
                              hipStream_t stream) {
  const float* heat = (const float*)d_in[0];
  const float* fm   = (const float*)d_in[1];
  const float* prev = (const float*)d_in[2];
  const float* n1w  = (const float*)d_in[3];
  const float* n1b  = (const float*)d_in[4];
  const float* n2w  = (const float*)d_in[5];
  const float* n2b  = (const float*)d_in[6];
  const float* p1w  = (const float*)d_in[7];
  const float* p1b  = (const float*)d_in[8];
  const float* p2w  = (const float*)d_in[9];
  const float* p2b  = (const float*)d_in[10];
  const float* cw   = (const float*)d_in[11];
  const float* cb   = (const float*)d_in[12];
  const float* sw   = (const float*)d_in[13];
  const float* sb   = (const float*)d_in[14];
  float* out = (float*)d_out;

  // workspace carve-out (~1.05 MB)
  char* ws = (char*)d_ws;
  float* wsCandVal = (float*)ws;                              // 16*8192 f32
  int*   wsCandIdx = (int*)(ws + 16 * 8192 * 4);              // 16*8192 i32
  int*   wsRows    = (int*)(ws + 2 * 16 * 8192 * 4);          // 128 i32
  int*   wsCols    = wsRows + 128;                            // 128 i32
  float* wsW       = (float*)(wsCols + 128);                  // 128 f32
  float* wsPooled  = wsW + 128;                               // 128 f32

  k_nms_tile<<<dim3(32, 32, NMAPS), 256, 0, stream>>>(heat, wsCandVal, wsCandIdx);
  k_topk_final<<<NMAPS, 256, 0, stream>>>(wsCandVal, wsCandIdx, wsRows, wsCols, out);
  k_pooled_mean<<<CFEAT, 256, 0, stream>>>(fm, wsPooled);
  k_fused_embed<<<1, 128, 0, stream>>>(fm, prev, n1w, n1b, n2w, n2b,
                                       p1w, p1b, p2w, p2b, cw, cb, sw, sb,
                                       wsRows, wsCols, wsPooled, wsW, out);
  k_bias_add<<<dim3(64, NMAPS), 256, 0, stream>>>(heat, wsRows, wsCols, wsW, out);
}
